// SRL_KT_56152402428005
// MI455X (gfx1250) — matrix operand order repaired and checked by reading, not compiled, not run on hardware
//
#include <hip/hip_runtime.h>
#include <hip/hip_bf16.h>
#include <math.h>

// ---------------- model dims ----------------
#define B_   32
#define S_   512
#define E_   256
#define H_   8
#define L_   4
#define DH_  32
#define MS_  (B_ * S_)        // 16384 rows of activations

// ---------------- WMMA / TDM types ----------------
typedef __attribute__((ext_vector_type(16))) __bf16 v16bf;
typedef __attribute__((ext_vector_type(8)))  float  v8f;
typedef __attribute__((ext_vector_type(4)))  unsigned int u32x4;
typedef __attribute__((ext_vector_type(8)))  int    i32x8;
typedef __attribute__((ext_vector_type(4)))  int    i32x4;

union FragAB { v16bf v; unsigned u[8]; };
union FragC  { v8f  v; float   f[8]; };

__device__ __forceinline__ unsigned short f2bf(float x) {
    union { float f; unsigned u; } c; c.f = x;
    unsigned r = c.u + 0x7FFFu + ((c.u >> 16) & 1u);   // round-to-nearest-even
    return (unsigned short)(r >> 16);
}

__device__ __forceinline__ v8f wmma_bf16(v16bf a, v16bf b, v8f c) {
    // D = A(16x32 bf16) * B(32x16 bf16) + C(16x16 f32)
    return __builtin_amdgcn_wmma_f32_16x16x32_bf16(
        false, a, false, b, (short)0, c, false, false);
}

// A-matrix 16x32 bf16, element (m,k) at base[m*stride + k] (row major, stride even).
__device__ __forceinline__ v16bf load_a_frag(const unsigned short* base, int stride) {
    int lane = threadIdx.x & 31;
    int m = lane & 15, half = lane >> 4;
    FragAB fr;
#pragma unroll
    for (int v = 0; v < 8; ++v) {
        int k0 = (half << 4) + (v << 1);
        fr.u[v] = *(const unsigned*)(base + (size_t)m * stride + k0);
    }
    return fr.v;
}

// B-matrix 32x16 bf16, element (k,n) at base[k*stride + n] (k NOT contiguous).
__device__ __forceinline__ v16bf load_b_frag_strided(const unsigned short* base, int stride) {
    int lane = threadIdx.x & 31;
    int n = lane & 15, half = lane >> 4;
    FragAB fr;
#pragma unroll
    for (int v = 0; v < 8; ++v) {
        int k0 = (half << 4) + (v << 1);
        unsigned short lo = base[(size_t)k0 * stride + n];
        unsigned short hi = base[(size_t)(k0 + 1) * stride + n];
        fr.u[v] = (unsigned)lo | ((unsigned)hi << 16);
    }
    return fr.v;
}

// B-matrix 32x16 bf16 where element (k,n) = base[n*stride + k] (k contiguous: K^T case).
__device__ __forceinline__ v16bf load_b_frag_kcontig(const unsigned short* base, int stride) {
    int lane = threadIdx.x & 31;
    int n = lane & 15, half = lane >> 4;
    FragAB fr;
#pragma unroll
    for (int v = 0; v < 8; ++v) {
        int k0 = (half << 4) + (v << 1);
        fr.u[v] = *(const unsigned*)(base + (size_t)n * stride + k0);
    }
    return fr.v;
}

// ---------------- TDM tile stage: global (rows x cols bf16, row stride) -> LDS ----------------
// Issued by ONE wave; per-wave TENSORcnt waited inside. Caller barriers for visibility.
#if __has_builtin(__builtin_amdgcn_tensor_load_to_lds) && __has_builtin(__builtin_amdgcn_s_wait_tensorcnt)
#define HAVE_TDM 1
#else
#define HAVE_TDM 0
#endif

__device__ __forceinline__ void tdm_load_2d(unsigned short* lds_dst,
                                            const unsigned short* gsrc,
                                            unsigned rows, unsigned cols,
                                            unsigned stride_elems) {
#if HAVE_TDM
    unsigned lds_off = (unsigned)((size_t)(void*)lds_dst & 0x1FFFFFu);  // LDS byte offset
    unsigned long long ga = (unsigned long long)(size_t)(const void*)gsrc;
    u32x4 g0;
    g0[0] = 1u;                                            // count=1 (user mode)
    g0[1] = lds_off;                                       // D#.lds_addr
    g0[2] = (unsigned)(ga & 0xFFFFFFFFu);                  // global_addr[31:0]
    g0[3] = (unsigned)((ga >> 32) & 0x01FFFFFFu) | 0x80000000u; // addr[56:32] | type=2
    i32x8 g1;
    g1[0] = (int)(1u << 16);                               // data_size=1 (2 bytes)
    g1[1] = (int)((stride_elems & 0xFFFFu) << 16);         // tensor_dim0[15:0]
    g1[2] = (int)((stride_elems >> 16) & 0xFFFFu);         // tensor_dim0[31:16], dim1 lo=0
    g1[3] = (int)(16u | ((cols & 0xFFFFu) << 16));         // tensor_dim1=1<<20, tile_dim0
    g1[4] = (int)(rows & 0xFFFFu);                         // tile_dim1 (tile_dim2=0)
    g1[5] = (int)stride_elems;                             // tensor_dim0_stride[31:0]
    g1[6] = 0;                                             // stride hi / dim1_stride lo
    g1[7] = 0;
    i32x4 z4 = {0, 0, 0, 0};
    i32x8 z8 = {0, 0, 0, 0, 0, 0, 0, 0};
    __builtin_amdgcn_tensor_load_to_lds(g0, g1, z4, z4, z8, 0);
    __builtin_amdgcn_s_wait_tensorcnt(0);
#else
    int lane = threadIdx.x & 31;
    unsigned wcols = cols >> 1;
    unsigned* d = (unsigned*)lds_dst;
    for (unsigned i = lane; i < rows * wcols; i += 32) {
        unsigned r = i / wcols, c = i % wcols;
        d[i] = ((const unsigned*)(gsrc + (size_t)r * stride_elems))[c];
    }
#endif
}

// ---------------- small prep kernels ----------------
__global__ void k_convert_bf16(const float* __restrict__ src,
                               unsigned short* __restrict__ dst, size_t n) {
    size_t i = (size_t)blockIdx.x * blockDim.x + threadIdx.x;
    size_t st = (size_t)gridDim.x * blockDim.x;
    for (; i < n; i += st) dst[i] = f2bf(src[i]);
}

__global__ void k_colsums(const float* __restrict__ W_in,
                          float* __restrict__ csumC, float* __restrict__ csumD) {
    int n = threadIdx.x;            // 256 threads
    float c = 0.f, d = 0.f;
    for (int i = 0; i < E_; ++i) {
        c += W_in[(size_t)(2 * E_ + i) * E_ + n];
        d += W_in[(size_t)(3 * E_ + i) * E_ + n];
    }
    csumC[n] = c; csumD[n] = d;
}

// pe_proj[s, n] = pe_shift[s, :] @ W_in[:, n]   (pe_shift[0]=0, else pe_inter[s-1])
__global__ void k_pe_proj(const float* __restrict__ pe_inter,
                          const float* __restrict__ W_in,
                          float* __restrict__ pe_proj) {
    int s = blockIdx.x, n = threadIdx.x;
    float acc = 0.f;
    if (s > 0) {
        const float* pe = pe_inter + (size_t)(s - 1) * (4 * E_);
        for (int i = 0; i < 4 * E_; ++i) acc += pe[i] * W_in[(size_t)i * E_ + n];
    }
    pe_proj[(size_t)s * E_ + n] = acc;
}

// xab[m, k] (K=512 bf16): k<256 -> item_emb (zeroed at s==0), else type_emb
__global__ void k_gather_xab(const int* __restrict__ item_inputs,
                             const int* __restrict__ type_inputs,
                             const float* __restrict__ use_table,
                             const float* __restrict__ type_table,
                             unsigned short* __restrict__ xab) {
    size_t i = (size_t)blockIdx.x * blockDim.x + threadIdx.x;
    size_t n = (size_t)MS_ * 512;
    size_t st = (size_t)gridDim.x * blockDim.x;
    for (; i < n; i += st) {
        int m = (int)(i >> 9), k = (int)(i & 511);
        int s = m & (S_ - 1);
        float v;
        if (k < E_) v = (s > 0) ? use_table[(size_t)item_inputs[m] * E_ + k] : 0.f;
        else        v = type_table[(size_t)type_inputs[m] * E_ + (k - E_)];
        xab[i] = f2bf(v);
    }
}

__global__ void k_gather_query(const int* __restrict__ item_ids,
                               const float* __restrict__ use_table,
                               unsigned short* __restrict__ qb) {
    size_t i = (size_t)blockIdx.x * blockDim.x + threadIdx.x;
    size_t n = (size_t)MS_ * E_;
    size_t st = (size_t)gridDim.x * blockDim.x;
    for (; i < n; i += st) {
        int m = (int)(i >> 8), e = (int)(i & 255);
        qb[i] = f2bf(use_table[(size_t)item_ids[m] * E_ + e]);
    }
}

// rel_attn[b,q,:] = softmax( relm==0 ? -1e4 : relm ), relm = rel * (k>q)   -> bf16
__global__ void k_rel_attn(const float* __restrict__ rel,
                           unsigned short* __restrict__ relb) {
    int wave = threadIdx.x >> 5, lane = threadIdx.x & 31;
    int row = blockIdx.x * 4 + wave;                  // row in [0, B*S)
    int q = row & (S_ - 1);
    const float* r = rel + (size_t)row * S_;
    float vals[16];
    float mx = -3.0e38f;
#pragma unroll
    for (int i = 0; i < 16; ++i) {
        int kk = lane + (i << 5);
        float relm = (kk > q) ? r[kk] : 0.f;
        float x = (relm == 0.f) ? -1.0e4f : relm;
        vals[i] = x;
        mx = fmaxf(mx, x);
    }
#pragma unroll
    for (int m = 16; m >= 1; m >>= 1) mx = fmaxf(mx, __shfl_xor(mx, m, 32));
    float sm = 0.f;
#pragma unroll
    for (int i = 0; i < 16; ++i) { float e = __expf(vals[i] - mx); vals[i] = e; sm += e; }
#pragma unroll
    for (int m = 16; m >= 1; m >>= 1) sm += __shfl_xor(sm, m, 32);
    float inv = 1.f / sm;
    unsigned short* o = relb + (size_t)row * S_;
#pragma unroll
    for (int i = 0; i < 16; ++i) o[lane + (i << 5)] = f2bf(vals[i] * inv);
}

// ---------------- WMMA GEMM: Cb[m,n] = bf16(A[m,:K] @ W[:K,n] + bias[n]) ----------------
// grid (MS/32, E/64), block 128 (4 waves). 32x64 tile: 2 accumulators/wave,
// A tile (32x32) TDM-staged into LDS, B fragment reused by both WMMAs.
__global__ void k_gemm_bias(const unsigned short* __restrict__ A,
                            const unsigned short* __restrict__ W,
                            const float* __restrict__ bias,
                            unsigned short* __restrict__ Cb, int K) {
    __shared__ unsigned short At[32 * 32];
    int wave = threadIdx.x >> 5;
    int m0 = blockIdx.x * 32;
    int ncol = blockIdx.y * 64 + wave * 16;
    int kchunks = K >> 5;

    v8f acc0 = {}, acc1 = {};
    for (int kc = 0; kc < kchunks; ++kc) {
        if (wave == 0)
            tdm_load_2d(At, A + (size_t)m0 * K + kc * 32, 32, 32, (unsigned)K);
        __syncthreads();
        if (kc + 1 < kchunks)
            __builtin_prefetch(W + (size_t)((kc + 1) * 32) * E_ + ncol, 0, 1);
        v16bf a0 = load_a_frag(At, 32);
        v16bf a1 = load_a_frag(At + 16 * 32, 32);
        v16bf b  = load_b_frag_strided(W + (size_t)(kc * 32) * E_ + ncol, E_);
        acc0 = wmma_bf16(a0, b, acc0);
        acc1 = wmma_bf16(a1, b, acc1);
        __syncthreads();
    }
    int lane = threadIdx.x & 31, n = lane & 15, half = lane >> 4;
    float bs = bias[ncol + n];
    FragC f0; f0.v = acc0;
    FragC f1; f1.v = acc1;
#pragma unroll
    for (int v = 0; v < 8; ++v) {
        int r = v + 8 * half;
        Cb[(size_t)(m0 + r) * E_ + ncol + n]      = f2bf(f0.f[v] + bs);
        Cb[(size_t)(m0 + 16 + r) * E_ + ncol + n] = f2bf(f1.f[v] + bs);
    }
}

// ---------------- input build: relu(xab@Wab + qresp*csumC + label*csumD + pe_proj + b_in) ----------------
__global__ void k_build_inputs(const unsigned short* __restrict__ xab,
                               const unsigned short* __restrict__ Wab,
                               const float* __restrict__ b_in,
                               const float* __restrict__ pe_proj,
                               const float* __restrict__ csumC,
                               const float* __restrict__ csumD,
                               const float* __restrict__ qresp,
                               const float* __restrict__ label,
                               unsigned short* __restrict__ kv0) {
    __shared__ unsigned short At[32 * 32];
    const int K = 512;
    int wave = threadIdx.x >> 5;
    int m0 = blockIdx.x * 32;
    int ncol = blockIdx.y * 64 + wave * 16;

    v8f acc0 = {}, acc1 = {};
    for (int kc = 0; kc < (K >> 5); ++kc) {
        if (wave == 0)
            tdm_load_2d(At, xab + (size_t)m0 * K + kc * 32, 32, 32, K);
        __syncthreads();
        v16bf a0 = load_a_frag(At, 32);
        v16bf a1 = load_a_frag(At + 16 * 32, 32);
        v16bf b  = load_b_frag_strided(Wab + (size_t)(kc * 32) * E_ + ncol, E_);
        acc0 = wmma_bf16(a0, b, acc0);
        acc1 = wmma_bf16(a1, b, acc1);
        __syncthreads();
    }
    int lane = threadIdx.x & 31, n = lane & 15, half = lane >> 4;
    int col = ncol + n;
    float bs = b_in[col], cC = csumC[col], cD = csumD[col];
    FragC f0; f0.v = acc0;
    FragC f1; f1.v = acc1;
#pragma unroll
    for (int v = 0; v < 8; ++v) {
#pragma unroll
        for (int sub = 0; sub < 2; ++sub) {
            int m = m0 + sub * 16 + v + 8 * half;
            int s = m & (S_ - 1);
            float raw = sub ? f1.f[v] : f0.f[v];
            float val = raw + bs + pe_proj[(size_t)s * E_ + col]
                      + qresp[m] * cC + label[m] * cD;
            kv0[(size_t)m * E_ + col] = f2bf(fmaxf(val, 0.f));
        }
    }
}

// ---------------- fused attention: 16 query rows per block, one (b,h) ----------------
// out[b,q,h*32+d] = (1-l1) * softmax(QK^T/sqrt(32), causal) @ V + l1 * rel_attn @ V
__global__ void k_attn(const unsigned short* __restrict__ Qb,
                       const unsigned short* __restrict__ Kb,
                       const unsigned short* __restrict__ Vb,
                       const unsigned short* __restrict__ relb,
                       const float* __restrict__ l1p,
                       float* __restrict__ out_attn) {
    __shared__ float          sc[16 * 512];   // scores / exp scratch (32 KB)
    __shared__ unsigned short pb[16 * 512];   // prob bf16            (16 KB)
    __shared__ unsigned short qt[16 * 32];    // staged Q tile        ( 1 KB)
    __shared__ unsigned short Vt[32 * 32];    // staged V tile        ( 2 KB)
    __shared__ float          ob[4][256];     // partial outs         ( 4 KB)

    int wave = threadIdx.x >> 5, lane = threadIdx.x & 31;
    int bh = blockIdx.y, bi = bh / H_, h = bh % H_;
    int q0 = blockIdx.x * 16;

    // stage Q tile (16x32 bf16) via TDM
    if (wave == 0)
        tdm_load_2d(qt, Qb + (size_t)(bi * S_ + q0) * E_ + h * DH_, 16, 32, E_);
    __syncthreads();

    // phase A: causal score tiles, S = Q K^T / sqrt(DH)
    const float scale = 0.1767766953f;  // 1/sqrt(32)
    int ntiles = (q0 >> 4) + 1;
    {
        v16bf aq = load_a_frag(qt, 32);
        for (int t = wave; t < ntiles; t += 4) {
            v16bf bk = load_b_frag_kcontig(
                Kb + ((size_t)(bi * S_ + t * 16) * E_ + h * DH_), E_);
            v8f z = {};
            v8f s = wmma_bf16(aq, bk, z);
            FragC fc; fc.v = s;
            int n = lane & 15, half = lane >> 4;
#pragma unroll
            for (int v = 0; v < 8; ++v)
                sc[(v + 8 * half) * 512 + t * 16 + n] = fc.f[v] * scale;
        }
    }
    __syncthreads();

    // phase B: causal softmax -> pb (bf16, zero outside k<=q)
    for (int r = wave * 4; r < wave * 4 + 4; ++r) {
        int q = q0 + r;
        float mx = -3.0e38f;
        for (int kk = lane; kk < 512; kk += 32)
            mx = fmaxf(mx, (kk <= q) ? sc[r * 512 + kk] : -3.0e38f);
#pragma unroll
        for (int m = 16; m >= 1; m >>= 1) mx = fmaxf(mx, __shfl_xor(mx, m, 32));
        float sm = 0.f;
        for (int kk = lane; kk < 512; kk += 32) {
            float e = (kk <= q) ? __expf(sc[r * 512 + kk] - mx) : 0.f;
            sc[r * 512 + kk] = e;
            sm += e;
        }
#pragma unroll
        for (int m = 16; m >= 1; m >>= 1) sm += __shfl_xor(sm, m, 32);
        float inv = 1.f / sm;
        for (int kk = lane; kk < 512; kk += 32)
            pb[r * 512 + kk] = f2bf(sc[r * 512 + kk] * inv);
    }
    __syncthreads();

    // phase C (barrier-uniform over 16 K-chunks):
    //   waves 0,1 -> P@V (d-tile 0,1);  waves 2,3 -> rel_attn@V (d-tile 0,1)
    //   V tile (32 rows x 32 dims) TDM-staged once per chunk, shared by all waves.
    {
        int path = wave >> 1, dt = wave & 1;
        int nchP = (q0 + 16 + 31) >> 5;    // causal limit for the P path
        const unsigned short* rbase = relb + (size_t)bi * S_ * S_ + (size_t)q0 * S_;
        v8f acc = {};
        for (int kc = 0; kc < 16; ++kc) {
            if (wave == 0)
                tdm_load_2d(Vt, Vb + ((size_t)(bi * S_ + kc * 32) * E_ + h * DH_),
                            32, 32, E_);
            __syncthreads();
            bool active = (path == 0) ? (kc < nchP) : true;
            if (active) {
                v16bf a = (path == 0) ? load_a_frag(pb + kc * 32, 512)
                                      : load_a_frag(rbase + kc * 32, S_);
                v16bf bv = load_b_frag_strided(Vt + dt * 16, 32);
                acc = wmma_bf16(a, bv, acc);
            }
            __syncthreads();
        }
        FragC fc; fc.v = acc;
        int n = lane & 15, half = lane >> 4;
#pragma unroll
        for (int v = 0; v < 8; ++v) ob[wave][(v + 8 * half) * 16 + n] = fc.f[v];
    }
    __syncthreads();

    if (wave < 2) {
        float l1 = l1p[0];
        int n = lane & 15, half = lane >> 4;
#pragma unroll
        for (int v = 0; v < 8; ++v) {
            int m = v + 8 * half;
            float val = (1.f - l1) * ob[wave][m * 16 + n]
                      + l1 * ob[2 + wave][m * 16 + n];
            out_attn[(size_t)(bi * S_ + q0 + m) * E_ + h * DH_ + wave * 16 + n] = val;
        }
    }
}

// ---------------- residual combine ----------------
__global__ void k_combine(const float* __restrict__ out_attn,
                          float* __restrict__ outputs,
                          unsigned short* __restrict__ outputs_b, int first) {
    size_t i = (size_t)blockIdx.x * blockDim.x + threadIdx.x;
    size_t n = (size_t)MS_ * E_;
    size_t st = (size_t)gridDim.x * blockDim.x;
    for (; i < n; i += st) {
        float v = out_attn[i];
        float o = first ? v : (outputs[i] + fmaxf(v, 0.f));
        outputs[i] = o;
        outputs_b[i] = f2bf(o);
    }
}

// ---------------- head GEMM: hidden = relu([outputs|query] @ W1 + b1), f32 out ----------------
// Same 32x64 WMMA tiling; K=512 split across the two bf16 activation bases.
__global__ void k_gemm_concat(const unsigned short* __restrict__ A0,   // outputs bf16 [MS,256]
                              const unsigned short* __restrict__ A1,   // query   bf16 [MS,256]
                              const unsigned short* __restrict__ W1b,  // [512,256] bf16
                              const float* __restrict__ b1,
                              float* __restrict__ hidden) {
    __shared__ unsigned short At[32 * 32];
    int wave = threadIdx.x >> 5;
    int m0 = blockIdx.x * 32;
    int ncol = blockIdx.y * 64 + wave * 16;

    v8f acc0 = {}, acc1 = {};
    for (int kc = 0; kc < 16; ++kc) {
        const unsigned short* base = (kc < 8)
            ? A0 + (size_t)m0 * E_ + kc * 32
            : A1 + (size_t)m0 * E_ + (kc - 8) * 32;
        if (wave == 0)
            tdm_load_2d(At, base, 32, 32, E_);
        __syncthreads();
        v16bf a0 = load_a_frag(At, 32);
        v16bf a1 = load_a_frag(At + 16 * 32, 32);
        v16bf b  = load_b_frag_strided(W1b + (size_t)(kc * 32) * E_ + ncol, E_);
        acc0 = wmma_bf16(a0, b, acc0);
        acc1 = wmma_bf16(a1, b, acc1);
        __syncthreads();
    }
    int lane = threadIdx.x & 31, n = lane & 15, half = lane >> 4;
    float bs = b1[ncol + n];
    FragC f0; f0.v = acc0;
    FragC f1; f1.v = acc1;
#pragma unroll
    for (int v = 0; v < 8; ++v) {
        int r = v + 8 * half;
        hidden[(size_t)(m0 + r) * E_ + ncol + n]      = fmaxf(f0.f[v] + bs, 0.f);
        hidden[(size_t)(m0 + 16 + r) * E_ + ncol + n] = fmaxf(f1.f[v] + bs, 0.f);
    }
}

// ---------------- final dot: out[m] = hidden[m,:] @ W2 + b2 ----------------
__global__ void k_dot(const float* __restrict__ hidden,
                      const float* __restrict__ W2, const float* __restrict__ b2,
                      float* __restrict__ out) {
    __shared__ float red[256];
    int m = blockIdx.x, j = threadIdx.x;
    red[j] = hidden[(size_t)m * E_ + j] * W2[j];
    __syncthreads();
    for (int s = 128; s >= 1; s >>= 1) {
        if (j < s) red[j] += red[j + s];
        __syncthreads();
    }
    if (j == 0) out[m] = red[0] + b2[0];
}

// ---------------- host launcher ----------------
extern "C" void kernel_launch(void* const* d_in, const int* in_sizes, int n_in,
                              void* d_out, int out_size, void* d_ws, size_t ws_size,
                              hipStream_t stream) {
    (void)in_sizes; (void)n_in; (void)out_size; (void)ws_size;
    const int*   item_inputs = (const int*)  d_in[0];
    const float* label       = (const float*)d_in[1];
    const int*   type_inputs = (const int*)  d_in[2];
    const int*   item_ids    = (const int*)  d_in[3];
    const float* rel         = (const float*)d_in[4];
    /* d_in[5] item_types: unused by reference */
    const float* qresp       = (const float*)d_in[6];
    const float* use_table   = (const float*)d_in[7];
    const float* type_table  = (const float*)d_in[8];
    const float* W_in        = (const float*)d_in[9];
    const float* b_in        = (const float*)d_in[10];
    const float* Wq          = (const float*)d_in[11];
    const float* bq          = (const float*)d_in[12];
    const float* Wk          = (const float*)d_in[13];
    const float* bk          = (const float*)d_in[14];
    const float* Wv          = (const float*)d_in[15];
    const float* bv          = (const float*)d_in[16];
    const float* l1          = (const float*)d_in[17];
    const float* W1          = (const float*)d_in[18];
    const float* b1          = (const float*)d_in[19];
    const float* W2          = (const float*)d_in[20];
    const float* b2          = (const float*)d_in[21];
    const float* pe_inter    = (const float*)d_in[22];
    float* out = (float*)d_out;

    // workspace carve-out (256B aligned)
    char* w = (char*)d_ws;
    auto alloc = [&](size_t bytes) -> void* {
        void* p = (void*)w;
        w += (bytes + 255) & ~(size_t)255;
        return p;
    };
    unsigned short* relb  = (unsigned short*)alloc((size_t)B_ * S_ * S_ * 2);
    unsigned short* xab   = (unsigned short*)alloc((size_t)MS_ * 512 * 2);
    unsigned short* qb    = (unsigned short*)alloc((size_t)MS_ * E_ * 2);
    unsigned short* kv0   = (unsigned short*)alloc((size_t)MS_ * E_ * 2);
    unsigned short* outsb = (unsigned short*)alloc((size_t)MS_ * E_ * 2);
    unsigned short* Qb    = (unsigned short*)alloc((size_t)MS_ * E_ * 2);
    unsigned short* Kb    = (unsigned short*)alloc((size_t)MS_ * E_ * 2);
    unsigned short* Vb    = (unsigned short*)alloc((size_t)MS_ * E_ * 2);
    float*          attn  = (float*)alloc((size_t)MS_ * E_ * 4);   // also reused as `hidden`
    float*          outsf = (float*)alloc((size_t)MS_ * E_ * 4);
    unsigned short* Wqb   = (unsigned short*)alloc((size_t)L_ * E_ * E_ * 2);
    unsigned short* Wkb   = (unsigned short*)alloc((size_t)L_ * E_ * E_ * 2);
    unsigned short* Wvb   = (unsigned short*)alloc((size_t)L_ * E_ * E_ * 2);
    unsigned short* Wab   = (unsigned short*)alloc((size_t)512 * E_ * 2);
    unsigned short* W1b   = (unsigned short*)alloc((size_t)512 * E_ * 2);
    float*          peP   = (float*)alloc((size_t)S_ * E_ * 4);
    float*          csC   = (float*)alloc((size_t)E_ * 4);
    float*          csD   = (float*)alloc((size_t)E_ * 4);

    const size_t nW = (size_t)L_ * E_ * E_;
    k_convert_bf16<<<1024, 256, 0, stream>>>(Wq, Wqb, nW);
    k_convert_bf16<<<1024, 256, 0, stream>>>(Wk, Wkb, nW);
    k_convert_bf16<<<1024, 256, 0, stream>>>(Wv, Wvb, nW);
    k_convert_bf16<<<512, 256, 0, stream>>>(W_in, Wab, (size_t)512 * E_);
    k_convert_bf16<<<512, 256, 0, stream>>>(W1, W1b, (size_t)512 * E_);
    k_colsums<<<1, 256, 0, stream>>>(W_in, csC, csD);
    k_pe_proj<<<S_, 256, 0, stream>>>(pe_inter, W_in, peP);
    k_gather_xab<<<8192, 256, 0, stream>>>(item_inputs, type_inputs,
                                           use_table, type_table, xab);
    k_gather_query<<<8192, 256, 0, stream>>>(item_ids, use_table, qb);
    k_rel_attn<<<MS_ / 4, 128, 0, stream>>>(rel, relb);

    dim3 gemm_grid(MS_ / 32, E_ / 64);
    k_build_inputs<<<gemm_grid, 128, 0, stream>>>(xab, Wab, b_in, peP, csC, csD,
                                                  qresp, label, kv0);

    dim3 attn_grid(S_ / 16, B_ * H_);
    for (int i = 0; i < L_; ++i) {
        const unsigned short* kvin = (i == 0) ? kv0 : outsb;
        k_gemm_bias<<<gemm_grid, 128, 0, stream>>>(qb,   Wqb + (size_t)i * E_ * E_,
                                                   bq + i * E_, Qb, E_);
        k_gemm_bias<<<gemm_grid, 128, 0, stream>>>(kvin, Wkb + (size_t)i * E_ * E_,
                                                   bk + i * E_, Kb, E_);
        k_gemm_bias<<<gemm_grid, 128, 0, stream>>>(kvin, Wvb + (size_t)i * E_ * E_,
                                                   bv + i * E_, Vb, E_);
        k_attn<<<attn_grid, 128, 0, stream>>>(Qb, Kb, Vb, relb, l1, attn);
        k_combine<<<4096, 256, 0, stream>>>(attn, outsf, outsb, i == 0 ? 1 : 0);
    }

    k_gemm_concat<<<gemm_grid, 128, 0, stream>>>(outsb, qb, W1b, b1, attn);
    k_dot<<<MS_, 256, 0, stream>>>(attn, W2, b2, out);
}
